// ComplexLiGRU_63488206569510
// MI455X (gfx1250) — matrix operand order repaired and checked by reading, not compiled, not run on hardware
//
#include <hip/hip_runtime.h>
#include <hip/hip_bf16.h>

typedef __attribute__((ext_vector_type(16))) __bf16 v16bf;
typedef __attribute__((ext_vector_type(8)))  __bf16 v8bf;
typedef __attribute__((ext_vector_type(8)))  float  v8f;
typedef __attribute__((ext_vector_type(4)))  int    v4i;

#define TQ 528   // padded LDS row stride (bf16 elems): 1056 B, 16B aligned, bank-skewed

// Optional CDNA5 async global->LDS copy path (probe via __has_builtin so the
// file compiles on toolchains that don't declare these builtins).
#if __has_builtin(__builtin_amdgcn_global_load_async_to_lds_b128) && \
    __has_builtin(__builtin_amdgcn_s_wait_asynccnt)
#define HAVE_ASYNC_LDS 1
#else
#define HAVE_ASYNC_LDS 0
#endif

// Builtin signature (from compiler diagnostic): param0 is global int4*, 4 args.
#define GAS(p) ((__attribute__((address_space(1))) v4i*)(p))
#define LAS(p) ((__attribute__((address_space(3))) v4i*)(p))

// Copy 16 bytes global(bf16) -> LDS(bf16), async when available.
__device__ __forceinline__ void copy16_g2l(__bf16* ldst, const __bf16* gsrc) {
#if HAVE_ASYNC_LDS
  __builtin_amdgcn_global_load_async_to_lds_b128(GAS(gsrc), LAS(ldst), 0, 0);
#else
  *(v8bf*)ldst = *(const v8bf*)gsrc;
#endif
}
__device__ __forceinline__ void copy_wait() {
#if HAVE_ASYNC_LDS
  __builtin_amdgcn_s_wait_asynccnt(0);
#endif
}

// Logical element of the complex block matrix [[Wr, Wi], [-Wi, Wr]] (512x1024)
__device__ __forceinline__ float big_elem(const float* __restrict__ Wr,
                                          const float* __restrict__ Wi,
                                          int k, int n) {
  if (n < 512) return (k < 256) ? Wr[k * 512 + n] : -Wi[(k - 256) * 512 + n];
  int nn = n - 512;
  return (k < 256) ? Wi[k * 512 + nn] : Wr[(k - 256) * 512 + nn];
}

// ---------------------------------------------------------------------------
// Kernel 0: build *fragment-swizzled* bf16 weight images, pre-convert x to
// bf16, zero h ping-pong + barrier counters.
// Swizzled layout: element ((nt*16+kc)*32 + lane)*16 + i  holds
//   Wbig[k = kc*32 + (lane>>4)*16 + i][n = nt*16 + (lane&15)]
// so one lane's B-fragment is 32 contiguous bytes (two b128 loads).
// ---------------------------------------------------------------------------
__global__ void __launch_bounds__(256) ligru_prep_kernel(
    const float* __restrict__ x,
    const float* __restrict__ Ww_r, const float* __restrict__ Ww_i,
    const float* __restrict__ Wu_r, const float* __restrict__ Wu_i,
    __bf16* __restrict__ Wswz, __bf16* __restrict__ Uswz,
    __bf16* __restrict__ xb,
    __bf16* __restrict__ hb, unsigned* __restrict__ cnt)
{
  const int idx = blockIdx.x * blockDim.x + threadIdx.x;
  const int stride = gridDim.x * blockDim.x;

  for (int o = idx; o < 512 * 1024; o += stride) {
    int i    = o & 15;
    int lane = (o >> 4) & 31;
    int kc   = (o >> 9) & 15;
    int nt   = o >> 13;
    int k = kc * 32 + (lane >> 4) * 16 + i;
    int n = nt * 16 + (lane & 15);
    Wswz[o] = (__bf16)big_elem(Ww_r, Ww_i, k, n);
    Uswz[o] = (__bf16)big_elem(Wu_r, Wu_i, k, n);
  }
  for (int o = idx; o < 16000 * 512; o += stride) xb[o] = (__bf16)x[o];
  if (idx < 2 * 16 * 512) hb[idx] = (__bf16)0.0f;
  if (idx < 1000)         cnt[idx] = 0u;
}

// ---------------------------------------------------------------------------
// Kernel 1: fused  w = complexLN( x @ Wbig ), written transposed [t*16+b][1024]
// One block per 16-row M tile; 8 waves; wave w owns N columns [128w, 128w+128).
// ---------------------------------------------------------------------------
__global__ void __launch_bounds__(256) ligru_gemm_ln_kernel(
    const __bf16* __restrict__ xb,        // [16000][512] bf16
    const __bf16* __restrict__ Wswz,      // fragment-swizzled [512x1024] bf16
    const float* __restrict__ g_rr, const float* __restrict__ g_ri,
    const float* __restrict__ g_ii, const float* __restrict__ beta,
    float* __restrict__ wout)             // [1000*16][1024], row = t*16 + b
{
  __shared__ __align__(16) char smem[16 * 1024 * 4];   // 64 KB, phase-aliased
  __bf16* xs  = (__bf16*)smem;                         // [16][TQ] during K loop
  float*  acc = (float*)smem;                          // [16][1024] after K loop

  const int tid   = threadIdx.x;
  const int wave  = tid >> 5;
  const int lane  = tid & 31;
  const int lhalf = lane >> 4;
  const int ln16  = lane & 15;
  const int m0    = blockIdx.x * 16;

  // Stage x tile (16 x 512 bf16 = 16 KB) via 16B chunks (async when available)
  for (int ch = tid; ch < 1024; ch += 256) {
    int r = ch >> 6, e = (ch & 63) * 8;                // 8 bf16 per 16B chunk
    copy16_g2l(&xs[r * TQ + e], &xb[(size_t)(m0 + r) * 512 + e]);
  }
  copy_wait();
  __syncthreads();

  v8f c[8];
#pragma unroll
  for (int nt = 0; nt < 8; ++nt) c[nt] = (v8f){0.f,0.f,0.f,0.f,0.f,0.f,0.f,0.f};

  const int ntile0 = wave * 8;

  for (int kc = 0; kc < 16; ++kc) {
    union { v16bf v; v8bf h2[2]; } A;
    const __bf16* ap = &xs[ln16 * TQ + kc * 32 + lhalf * 16];
    A.h2[0] = *(const v8bf*)ap;
    A.h2[1] = *(const v8bf*)(ap + 8);
#pragma unroll
    for (int nt = 0; nt < 8; ++nt) {
      // swizzled fragment: 32 contiguous bytes per lane
      const __bf16* bp =
          &Wswz[(((size_t)(ntile0 + nt) * 16 + kc) * 32 + lane) * 16];
      union { v16bf v; v8bf h2[2]; } B;
      B.h2[0] = *(const v8bf*)bp;
      B.h2[1] = *(const v8bf*)(bp + 8);
      c[nt] = __builtin_amdgcn_wmma_f32_16x16x32_bf16(
          false, A.v, false, B.v, (short)0, c[nt], false, false);
    }
  }
  __syncthreads();   // done with xs; reuse smem as fp32 accumulator

#pragma unroll
  for (int nt = 0; nt < 8; ++nt)
#pragma unroll
    for (int v = 0; v < 8; ++v)
      acc[(v + 8 * lhalf) * 1024 + (ntile0 + nt) * 16 + ln16] = c[nt][v];
  __syncthreads();

  // Complex layernorm: wave handles rows {wave, wave+8}
  for (int rr = 0; rr < 2; ++rr) {
    const int r = wave + 8 * rr;
    const float* row = &acc[r * 1024];
    float sr = 0.f, si = 0.f, srr = 0.f, sii = 0.f, sri = 0.f;
    for (int j = lane; j < 512; j += 32) {
      float a = row[j], b = row[j + 512];
      sr += a; si += b; srr += a * a; sii += b * b; sri += a * b;
    }
#pragma unroll
    for (int off = 16; off >= 1; off >>= 1) {
      sr  += __shfl_xor(sr,  off, 32);
      si  += __shfl_xor(si,  off, 32);
      srr += __shfl_xor(srr, off, 32);
      sii += __shfl_xor(sii, off, 32);
      sri += __shfl_xor(sri, off, 32);
    }
    const float inv = 1.0f / 512.0f;
    float mr  = sr * inv, mi = si * inv;
    float Vrr = srr * inv - mr * mr + 1e-4f;
    float Vii = sii * inv - mi * mi + 1e-4f;
    float Vri = sri * inv - mr * mi;
    float tau = Vrr + Vii;
    float dlt = Vrr * Vii - Vri * Vri;
    float s   = sqrtf(dlt);
    float tt  = sqrtf(tau + 2.0f * s);
    float ist = 1.0f / (s * tt);
    float Wrr = (Vii + s) * ist, Wii = (Vrr + s) * ist, Wri = -Vri * ist;

    const int m = m0 + r;
    const int b = m / 1000;
    const int t = m - b * 1000;
    float* orow = &wout[(size_t)(t * 16 + b) * 1024];
    for (int j = lane; j < 512; j += 32) {
      float wrc = row[j] - mr, wic = row[j + 512] - mi;
      float xr = Wrr * wrc + Wri * wic;
      float xi = Wri * wrc + Wii * wic;
      orow[j]       = g_rr[j] * xr + g_ri[j] * xi + beta[j];
      orow[j + 512] = g_ri[j] * xr + g_ii[j] * xi + beta[j + 512];
    }
  }
}

// ---------------------------------------------------------------------------
// Kernel 2: persistent LiGRU scan. 8 blocks x 8 waves; block k owns h columns
// [64k, 64k+64) (gr cols 64k.. and gi cols 512+64k..). Wave w owns one 16-wide
// N tile; its 16 B fragments stay in VGPRs for all 1000 steps. h ping-pongs
// through global as bf16; staged to LDS each step (async path when available).
// ---------------------------------------------------------------------------
__global__ void __launch_bounds__(256) ligru_scan_kernel(
    const __bf16* __restrict__ Uswz,   // fragment-swizzled [512x1024] bf16
    const float* __restrict__ wpre,    // [1000*16][1024]
    __bf16* hb,                        // [2][16][512] bf16 ping-pong
    unsigned* cnt,                     // [1000] barrier counters (pre-zeroed)
    float* __restrict__ out,           // [16][1000][512]
    int nblocks)
{
  __shared__ __align__(16) __bf16 hs[16 * TQ];   // staged h (bf16)
  __shared__ float gbuf[16 * 128];               // this block's gate outputs
  __shared__ float hloc[16 * 64];                // fp32 local h slice

  const int tid   = threadIdx.x;
  const int wave  = tid >> 5;
  const int lane  = tid & 31;
  const int lhalf = lane >> 4;
  const int ln16  = lane & 15;
  const int blk   = blockIdx.x;

  const int c0 = (wave < 4) ? (blk * 64 + wave * 16)
                            : (512 + blk * 64 + (wave - 4) * 16);
  const int ntile = c0 >> 4;

  // Preload loop-invariant B fragments into registers (16 x v16bf = 128 VGPRs)
  v16bf Bfrag[16];
#pragma unroll
  for (int kc = 0; kc < 16; ++kc) {
    const __bf16* bp = &Uswz[(((size_t)ntile * 16 + kc) * 32 + lane) * 16];
    union { v16bf v; v8bf h2[2]; } B;
    B.h2[0] = *(const v8bf*)bp;
    B.h2[1] = *(const v8bf*)(bp + 8);
    Bfrag[kc] = B.v;
  }

  for (int i = tid; i < 16 * 64; i += 256) hloc[i] = 0.0f;

  for (int t = 0; t < 1000; ++t) {
    // Stage full h (16x512 bf16 = 16 KB) from ping-pong slot t&1
    const __bf16* hin = hb + (size_t)(t & 1) * 16 * 512;
    for (int ch = tid; ch < 1024; ch += 256) {
      int r = ch >> 6, e = (ch & 63) * 8;
      copy16_g2l(&hs[r * TQ + e], &hin[r * 512 + e]);
    }
    copy_wait();
    __syncthreads();

    v8f c = {0.f,0.f,0.f,0.f,0.f,0.f,0.f,0.f};
#pragma unroll
    for (int kc = 0; kc < 16; ++kc) {
      union { v16bf v; v8bf h2[2]; } A;
      const __bf16* ap = &hs[ln16 * TQ + kc * 32 + lhalf * 16];
      A.h2[0] = *(const v8bf*)ap;
      A.h2[1] = *(const v8bf*)(ap + 8);
      c = __builtin_amdgcn_wmma_f32_16x16x32_bf16(
          false, A.v, false, Bfrag[kc], (short)0, c, false, false);
    }

    // gates = w_t + h @ Ubig
    const float* wrow = &wpre[(size_t)t * 16 * 1024];
#pragma unroll
    for (int v = 0; v < 8; ++v)
      c[v] += wrow[(size_t)(v + 8 * lhalf) * 1024 + c0 + ln16];

    if (t + 1 < 1000)  // hint next step's w_t block into cache
      __builtin_prefetch(&wpre[(size_t)(t + 1) * 16 * 1024 + (size_t)tid * 64], 0, 1);

    const int cl = (wave < 4) ? (wave * 16 + ln16) : (64 + (wave - 4) * 16 + ln16);
#pragma unroll
    for (int v = 0; v < 8; ++v)
      gbuf[(v + 8 * lhalf) * 128 + cl] = c[v];
    __syncthreads();

    // h update for this block's 64 columns (fp32 state path via hloc)
    __bf16* hout = hb + (size_t)((t + 1) & 1) * 16 * 512;
    for (int i = tid; i < 16 * 64; i += 256) {
      int r = i >> 6, jj = i & 63;
      float a  = gbuf[r * 128 + jj];
      float zp = gbuf[r * 128 + 64 + jj];
      float z  = 1.0f / (1.0f + __expf(-zp));
      float hn = z * hloc[i] + (1.0f - z) * fmaxf(a, 0.0f);
      hloc[i] = hn;
      hout[r * 512 + blk * 64 + jj] = (__bf16)hn;
      out[(size_t)r * 512000 + (size_t)t * 512 + blk * 64 + jj] = hn;
    }
    __threadfence();
    __syncthreads();

    // Cross-workgroup barrier: cluster barrier (NOP outside a cluster) +
    // per-step atomic counter (pre-zeroed each launch -> replay-safe).
    __builtin_amdgcn_s_cluster_barrier();
    if (tid == 0) {
      atomicAdd(&cnt[t], 1u);
      while (atomicAdd(&cnt[t], 0u) < (unsigned)nblocks)
        __builtin_amdgcn_s_sleep(2);
    }
    __syncthreads();
    __threadfence();
  }
}

// ---------------------------------------------------------------------------
extern "C" void kernel_launch(void* const* d_in, const int* in_sizes, int n_in,
                              void* d_out, int out_size, void* d_ws, size_t ws_size,
                              hipStream_t stream) {
  const float* x    = (const float*)d_in[0];
  const float* Ww_r = (const float*)d_in[1];
  const float* Ww_i = (const float*)d_in[2];
  const float* Wu_r = (const float*)d_in[3];
  const float* Wu_i = (const float*)d_in[4];
  const float* g_rr = (const float*)d_in[5];
  const float* g_ri = (const float*)d_in[6];
  const float* g_ii = (const float*)d_in[7];
  const float* beta = (const float*)d_in[8];

  char* ws = (char*)d_ws;
  size_t off = 0;
  __bf16*   Wswz = (__bf16*)(ws + off); off += (size_t)1 << 20;          // 1 MB
  __bf16*   Uswz = (__bf16*)(ws + off); off += (size_t)1 << 20;          // 1 MB
  float*    wpre = (float*)(ws + off);  off += (size_t)16000 * 1024 * 4; // 62.5 MB
  __bf16*   hb   = (__bf16*)(ws + off); off += 2 * 16 * 512 * 2;         // 32 KB
  unsigned* cnt  = (unsigned*)(ws + off); off += 4096;
  __bf16*   xb   = (__bf16*)(ws + off); off += (size_t)16000 * 512 * 2;  // 16 MB

  ligru_prep_kernel<<<2048, 256, 0, stream>>>(x, Ww_r, Ww_i, Wu_r, Wu_i,
                                              Wswz, Uswz, xb, hb, cnt);
  ligru_gemm_ln_kernel<<<1000, 256, 0, stream>>>(xb, Wswz, g_rr, g_ri, g_ii,
                                                 beta, wpre);
  ligru_scan_kernel<<<8, 256, 0, stream>>>(Uswz, wpre, hb, cnt,
                                           (float*)d_out, 8);
}
